// NADEInitializationPolicy_4054449127518
// MI455X (gfx1250) — matrix operand order repaired and checked by reading, not compiled, not run on hardware
//
#include <hip/hip_runtime.h>
#include <hip/hip_bf16.h>
#include <stdint.h>

#define DP 2048
#define DC 1024
#define DH 8192
#define WSTRIDE (DC + DP)   // 3072, row stride of W

typedef __attribute__((ext_vector_type(16))) _Float16 v16h;
typedef __attribute__((ext_vector_type(8)))  float    v8f;

#if defined(__HIP_DEVICE_COMPILE__) && defined(__gfx1250__)
#define HAS_ASYNC_LDS 1
#else
#define HAS_ASYNC_LDS 0
#endif

#if HAS_ASYNC_LDS
// copy 16 bytes (4 floats) global -> LDS via the async unit (ASYNCcnt).
// VFLAT async encoding: VDST = per-lane LDS byte offset (low 32 bits of the
// generic pointer, per ISA flat-aperture rule), VADDR = 64-bit global addr.
__device__ __forceinline__ void async_cp16(const float* g, float* l) {
  const uint32_t lds = (uint32_t)(uintptr_t)l;
  const uint64_t ga  = (uint64_t)(uintptr_t)g;
  asm volatile("global_load_async_to_lds_b128 %0, %1, off"
               :: "v"(lds), "v"(ga) : "memory");
}
__device__ __forceinline__ void wait_async_le2() {
  asm volatile("s_wait_asynccnt 0x2" ::: "memory");
}
#endif

// ---------------------------------------------------------------------------
// Kernel 1: a0 = c + W[:, :DC] @ context  via v_wmma_f32_16x16x32_f16.
// One wave per 16-row tile. B replicates the context chunk into all 16
// columns, so every lane's accumulator holds the GEMV result for its rows.
// f32 inputs split into hi/lo f16 halves; 3 WMMAs recover ~f32 accuracy.
// ---------------------------------------------------------------------------
__global__ void nade_a0_wmma(const float* __restrict__ W,
                             const float* __restrict__ ctx,
                             const float* __restrict__ c,
                             float* __restrict__ a0) {
  const int lane = threadIdx.x & 31;
  const int wave = (blockIdx.x * blockDim.x + threadIdx.x) >> 5; // 0..511
  const int m0   = wave * 16;
  const int hl   = lane >> 4;
  const int m    = m0 + (lane & 15);

  v8f acc = {};
  for (int kb = 0; kb < DC; kb += 32) {
    float aw[16], bw[16];
#pragma unroll
    for (int e = 0; e < 16; ++e) {
      const int ka = (e < 8) ? (kb + e + 8 * hl) : (kb + e + 8 + 8 * hl);
      const int kn = (e < 8) ? (kb + e + 8 * hl) : (kb + e + 8 + 8 * hl);
      aw[e] = W[(size_t)m * WSTRIDE + ka];
      bw[e] = ctx[kn];
    }
    v16h ahi, alo, bhi, blo;
#pragma unroll
    for (int e = 0; e < 16; ++e) {
      _Float16 h = (_Float16)aw[e];
      ahi[e] = h;  alo[e] = (_Float16)(aw[e] - (float)h);
      h = (_Float16)bw[e];
      bhi[e] = h;  blo[e] = (_Float16)(bw[e] - (float)h);
    }
    acc = __builtin_amdgcn_wmma_f32_16x16x32_f16(false, ahi, false, bhi,
                                                 (short)0, acc, false, false);
    acc = __builtin_amdgcn_wmma_f32_16x16x32_f16(false, alo, false, bhi,
                                                 (short)0, acc, false, false);
    acc = __builtin_amdgcn_wmma_f32_16x16x32_f16(false, ahi, false, blo,
                                                 (short)0, acc, false, false);
  }
  // D layout: VGPR r, lanes 0-15 -> M=r ; lanes 16-31 -> M=r+8; all N equal.
  if ((lane & 15) == 0) {
    const int base = m0 + 8 * hl;
#pragma unroll
    for (int r = 0; r < 8; ++r) a0[base + r] = acc[r] + c[base + r];
  }
}

// ---------------------------------------------------------------------------
// Kernel 2: transpose W[:, DC:]  ->  WT[DP][DH]  (coalesced both sides).
// ---------------------------------------------------------------------------
__global__ void nade_transposeW2(const float* __restrict__ W,
                                 float* __restrict__ WT) {
  __shared__ float tile[32][33];
  const int ib = blockIdx.x * 32;   // DP index
  const int jb = blockIdx.y * 32;   // DH index
  const int tx = threadIdx.x, ty = threadIdx.y;
#pragma unroll
  for (int r = 0; r < 32; r += 8)
    tile[ty + r][tx] = W[(size_t)(jb + ty + r) * WSTRIDE + DC + ib + tx];
  __syncthreads();
#pragma unroll
  for (int r = 0; r < 32; r += 8)
    WT[(size_t)(ib + ty + r) * DH + jb + tx] = tile[tx][ty + r];
}

// ---------------------------------------------------------------------------
// Exact JAX threefry2x32-20 block.
// ---------------------------------------------------------------------------
__device__ __forceinline__ uint32_t rotl32(uint32_t v, int r) {
  return (v << r) | (v >> (32 - r));
}
__device__ void tf_block(uint32_t k0, uint32_t k1, uint32_t x0, uint32_t x1,
                         uint32_t& o0, uint32_t& o1) {
  const uint32_t ks2 = 0x1BD11BDAu ^ k0 ^ k1;
  x0 += k0; x1 += k1;
#define TF_R4(a, b, c, d)                                                  \
  x0 += x1; x1 = rotl32(x1, a); x1 ^= x0;                                  \
  x0 += x1; x1 = rotl32(x1, b); x1 ^= x0;                                  \
  x0 += x1; x1 = rotl32(x1, c); x1 ^= x0;                                  \
  x0 += x1; x1 = rotl32(x1, d); x1 ^= x0;
  TF_R4(13, 15, 26, 6)  x0 += k1;  x1 += ks2 + 1u;
  TF_R4(17, 29, 16, 24) x0 += ks2; x1 += k0 + 2u;
  TF_R4(13, 15, 26, 6)  x0 += k0;  x1 += k1 + 3u;
  TF_R4(17, 29, 16, 24) x0 += k1;  x1 += ks2 + 4u;
  TF_R4(13, 15, 26, 6)  x0 += ks2; x1 += k0 + 5u;
#undef TF_R4
  o0 = x0; o1 = x1;
}
// bits for step i, replicating split(key(42), DP)[i] + scalar bernoulli.
__device__ uint32_t sample_bits(int i) {
  const uint32_t n = DP;
  const uint32_t idx = 2u * (uint32_t)i;
  const uint32_t j = (idx < n) ? idx : (idx - n);
  uint32_t o0a, o1a, o0b, o1b;
  tf_block(0u, 42u, j,      n + j,      o0a, o1a);
  tf_block(0u, 42u, j + 1u, n + j + 1u, o0b, o1b);
  const uint32_t key0 = (idx < n) ? o0a : o1a;
  const uint32_t key1 = (idx < n) ? o0b : o1b;
  uint32_t b0, b1;
  tf_block(key0, key1, 0u, 0u, b0, b1); // padded (0,0) block
  return b0;
}

// ---------------------------------------------------------------------------
// Kernel 3: sequential 2048-step loop. ONE workgroup (32 waves, one WGP);
// LDS barriers beat any cross-WGP sync per step. Thread t owns hidden units
// j = t*8 .. t*8+7, so each thread reads back exactly the bytes it staged
// -> wave-local s_wait_asynccnt is sufficient ordering, no extra barrier.
// V rows double-buffered in LDS via global_load_async_to_lds_b128.
// ---------------------------------------------------------------------------
__global__ __launch_bounds__(1024)
void nade_seq(const float* __restrict__ V, const float* __restrict__ b,
              const float* __restrict__ W, const float* __restrict__ WT,
              const float* __restrict__ a0, float* __restrict__ out,
              int useWT) {
  const int t = threadIdx.x;
  const int lane = t & 31, wid = t >> 5;

  __shared__ float red[32];
  __shared__ float xsh;
#if HAS_ASYNC_LDS
  __shared__ float vbuf[2][DH];     // 64KB double buffer (LDS is 320KB/WGP)
#endif

  float a[8], h[8];
#pragma unroll
  for (int k = 0; k < 8; ++k) {
    a[k] = a0[t * 8 + k];
    h[k] = 1.f / (1.f + __expf(-a[k]));
  }

#if HAS_ASYNC_LDS
  // stage rows 0 and 1
  async_cp16(V + t * 8,          &vbuf[0][t * 8]);
  async_cp16(V + t * 8 + 4,      &vbuf[0][t * 8 + 4]);
  async_cp16(V + DH + t * 8,     &vbuf[1][t * 8]);
  async_cp16(V + DH + t * 8 + 4, &vbuf[1][t * 8 + 4]);
#else
  float vcur[8];
#pragma unroll
  for (int k = 0; k < 8; ++k) vcur[k] = V[t * 8 + k];   // row 0
#endif

  float logp = 0.f;
  float xlast = 1.f;

  for (int i = 0; i < DP; ++i) {
    // -- L2 prefetch (no counters, free on the critical path) --
    if (t < 256 && i + 4 < DP)
      __builtin_prefetch(V + (size_t)(i + 4) * DH + t * 32, 0, 3);
    if (useWT && t >= 256 && t < 512 && i + 1 < DP)
      __builtin_prefetch(WT + (size_t)(i + 1) * DH + (t - 256) * 32, 0, 3);

    // -- register prefetch of WT row i (used only if x==1) --
    float wt[8];
#pragma unroll
    for (int k = 0; k < 8; ++k) {
      const int j = t * 8 + k;
      wt[k] = useWT ? WT[(size_t)i * DH + j]
                    : W[(size_t)j * WSTRIDE + DC + i];
    }

#if HAS_ASYNC_LDS
    wait_async_le2();                      // row i staged (in-order done)
    float vcur[8];
#pragma unroll
    for (int k = 0; k < 8; ++k) vcur[k] = vbuf[i & 1][t * 8 + k];
#else
    float vnx[8];
#pragma unroll
    for (int k = 0; k < 8; ++k)
      vnx[k] = (i + 1 < DP) ? V[(size_t)(i + 1) * DH + t * 8 + k] : 0.f;
#endif

    // -- h = sigmoid(a): only changed if last sample was 1 --
    if (xlast != 0.f) {
#pragma unroll
      for (int k = 0; k < 8; ++k) h[k] = 1.f / (1.f + __expf(-a[k]));
    }

    float partial = 0.f;
#pragma unroll
    for (int k = 0; k < 8; ++k) partial = fmaf(vcur[k], h[k], partial);

#if HAS_ASYNC_LDS
    // overwrite our own just-consumed region with row i+2
    if (i + 2 < DP) {
      const float* g = V + (size_t)(i + 2) * DH + t * 8;
      float* l = &vbuf[i & 1][t * 8];
      async_cp16(g, l);
      async_cp16(g + 4, l + 4);
    }
#endif

    // -- wave32 reduction, then cross-wave via LDS --
#pragma unroll
    for (int m = 16; m >= 1; m >>= 1) partial += __shfl_xor(partial, m, 32);
    if (lane == 0) red[wid] = partial;
    __syncthreads();

    if (t < 32) {
      float v = red[t];
#pragma unroll
      for (int m = 16; m >= 1; m >>= 1) v += __shfl_xor(v, m, 32);
      if (t == 0) {
        const float p = 1.f / (1.f + __expf(-(b[i] + v)));
        const uint32_t bits = sample_bits(i);
        const float u =
            __uint_as_float((bits >> 9) | 0x3f800000u) - 1.0f; // [0,1)
        const float x = (u < p) ? 1.f : 0.f;
        logp += (x != 0.f) ? logf(p) : log1pf(-p);
        out[i] = x;
        xsh = x;
      }
    }
    __syncthreads();

    const float x = xsh;
    if (x != 0.f) {
#pragma unroll
      for (int k = 0; k < 8; ++k) a[k] += wt[k];
    }
    xlast = x;
#if !HAS_ASYNC_LDS
#pragma unroll
    for (int k = 0; k < 8; ++k) vcur[k] = vnx[k];
#endif
  }
  if (t == 0) out[DP] = logp;
}

// ---------------------------------------------------------------------------
// inputs: [0]=context(1024) [1]=W(8192*3072) [2]=V(2048*8192) [3]=b(2048)
//         [4]=c(8192);  out: 2048 samples + 1 logp (f32).
// ws: [0,32KB) = a0 buffer; [32KB, 32KB+64MB) = W2 transposed (if it fits).
// ---------------------------------------------------------------------------
extern "C" void kernel_launch(void* const* d_in, const int* in_sizes, int n_in,
                              void* d_out, int out_size, void* d_ws,
                              size_t ws_size, hipStream_t stream) {
  (void)in_sizes; (void)n_in; (void)out_size;
  const float* ctx = (const float*)d_in[0];
  const float* W   = (const float*)d_in[1];
  const float* V   = (const float*)d_in[2];
  const float* b   = (const float*)d_in[3];
  const float* c   = (const float*)d_in[4];
  float* out = (float*)d_out;

  float* a0 = (float*)d_ws;
  const size_t wt_off = 32768;
  float* WT = (float*)((char*)d_ws + wt_off);
  const int useWT =
      (ws_size >= wt_off + (size_t)DP * DH * sizeof(float)) ? 1 : 0;

  nade_a0_wmma<<<dim3(64), dim3(256), 0, stream>>>(W, ctx, c, a0);
  if (useWT)
    nade_transposeW2<<<dim3(DP / 32, DH / 32), dim3(32, 8), 0, stream>>>(W, WT);
  nade_seq<<<dim3(1), dim3(1024), 0, stream>>>(V, b, W, useWT ? WT : nullptr,
                                               a0, out, useWT);
}
